// GAT_51384988730013
// MI455X (gfx1250) — compile-verified, matrix operand order checked by reading, not run on hardware
//
#include <hip/hip_runtime.h>
#include <math.h>

typedef __attribute__((ext_vector_type(16))) _Float16 v16h;
typedef __attribute__((ext_vector_type(8)))  float    v8f;

#define NEG_SLOPE 0.2f

// ---------------------------------------------------------------- utilities

__global__ void fill_kernel(float* __restrict__ p, float v, int n) {
  int i = blockIdx.x * blockDim.x + threadIdx.x;
  if (i < n) p[i] = v;
}

__device__ __forceinline__ void atomic_max_float(float* addr, float val) {
  // sign-aware: non-negative floats order like ints, negatives reverse-order as uints
  if (val >= 0.0f) atomicMax((int*)addr, __float_as_int(val));
  else             atomicMin((unsigned int*)addr, __float_as_uint(val));
}

__device__ __forceinline__ void edge_src_dst(const long long* __restrict__ ei,
                                             int E, int e, int& s, int& d) {
  if (e < E) { s = (int)ei[e]; d = (int)ei[E + e]; }
  else       { s = e - E; d = e - E; }   // self-loops appended
}

__device__ __forceinline__ float leaky(float x) {
  return x > 0.0f ? x : NEG_SLOPE * x;
}

// ---------------------------------------------------- dense GEMM via WMMA
// X:[n,FIN] row-major, Wm:[FIN,FOUT] row-major, H:[n,FOUT] row-major.
// One wave computes a 16x16 tile of H; 4 waves per block cover 4 row tiles.
// Requires n % 16 == 0 (N = 50000 = 3125*16). Guard is wave-uniform so EXEC
// stays all-ones across the WMMA ops (ISA 7.12 requirement).
// A layout (16x32 f16, ISA 7.12.2): lane L(0-15) holds row M=L, halves 0..7 =
// K {kb..kb+7}, halves 8..15 = K {kb+16..kb+23} with kb = (lane>>4)*8.
// B layout (32x16 f16): lane L holds column N=L&15 with the same K striping.
template <int FIN, int FOUT>
__global__ void gemm_wmma_kernel(const float* __restrict__ X,
                                 const float* __restrict__ Wm,
                                 float* __restrict__ H, int n) {
  const int lane    = threadIdx.x & 31;
  const int wave    = threadIdx.x >> 5;
  const int rowBase = (blockIdx.x * 4 + wave) * 16;
  const int colBase = blockIdx.y * 16;
  if (rowBase >= n) return;                 // wave-uniform
  const int halfSel = lane >> 4;            // 0 or 1
  const int kb      = halfSel * 8;
  const int col     = colBase + (lane & 15);
  const float* xrow = X + (rowBase + (lane & 15)) * FIN;

  v8f acc = {};
#pragma unroll
  for (int k0 = 0; k0 < FIN; k0 += 32) {
    // A: 8 + 8 contiguous K values per lane -> 4x float4 (global_load_b128)
    float4 a0 = *(const float4*)(xrow + k0 + kb);
    float4 a1 = *(const float4*)(xrow + k0 + kb + 4);
    float4 a2 = *(const float4*)(xrow + k0 + 16 + kb);
    float4 a3 = *(const float4*)(xrow + k0 + 16 + kb + 4);
    v16h a, b;
    a[0]  = (_Float16)a0.x; a[1]  = (_Float16)a0.y;
    a[2]  = (_Float16)a0.z; a[3]  = (_Float16)a0.w;
    a[4]  = (_Float16)a1.x; a[5]  = (_Float16)a1.y;
    a[6]  = (_Float16)a1.z; a[7]  = (_Float16)a1.w;
    a[8]  = (_Float16)a2.x; a[9]  = (_Float16)a2.y;
    a[10] = (_Float16)a2.z; a[11] = (_Float16)a2.w;
    a[12] = (_Float16)a3.x; a[13] = (_Float16)a3.y;
    a[14] = (_Float16)a3.z; a[15] = (_Float16)a3.w;
#pragma unroll
    for (int i = 0; i < 8; ++i) {
      b[i]     = (_Float16)Wm[(k0 + kb + i)      * FOUT + col];
      b[i + 8] = (_Float16)Wm[(k0 + 16 + kb + i) * FOUT + col];
    }
    acc = __builtin_amdgcn_wmma_f32_16x16x32_f16(
        /*neg_a=*/false, a, /*neg_b=*/false, b,
        /*c_mod=*/(short)0, acc, /*reuse_a=*/false, /*reuse_b=*/false);
  }
  // D layout: VGPR v -> row M = v + halfSel*8, col N = lane&15
  float* hp = H + (rowBase + halfSel * 8) * FOUT + col;
#pragma unroll
  for (int v = 0; v < 8; ++v) hp[v * FOUT] = acc[v];
}

// --------------------------------------------- per-node attention scores
__global__ void att_kernel(const float* __restrict__ H,
                           const float* __restrict__ attS,
                           const float* __restrict__ attD,
                           float* __restrict__ aS, float* __restrict__ aD,
                           int n, int heads, int ch) {
  int t = blockIdx.x * blockDim.x + threadIdx.x;
  if (t >= n * heads) return;
  int node = t / heads, hh = t - node * heads;
  const float* hp = H + node * heads * ch + hh * ch;
  const float* ps = attS + hh * ch;
  const float* pd = attD + hh * ch;
  float ss = 0.0f, sd = 0.0f;
  for (int c = 0; c < ch; ++c) {
    float v = hp[c];
    ss += v * ps[c];
    sd += v * pd[c];
  }
  aS[t] = ss;
  aD[t] = sd;
}

// -------------------------------------------------------- edge pass 1: max
__global__ void edge_max_kernel(const long long* __restrict__ ei, int E, int ET,
                                const float* __restrict__ aS,
                                const float* __restrict__ aD,
                                float* __restrict__ Mx, int heads) {
  int t = blockIdx.x * blockDim.x + threadIdx.x;
  if (t >= ET * heads) return;
  int e = t / heads, hh = t - e * heads;
  int s, d; edge_src_dst(ei, E, e, s, d);
  float l = leaky(aS[s * heads + hh] + aD[d * heads + hh]);
  atomic_max_float(&Mx[d * heads + hh], l);
}

// -------------------------------------------------------- edge pass 2: sum
__global__ void edge_sum_kernel(const long long* __restrict__ ei, int E, int ET,
                                const float* __restrict__ aS,
                                const float* __restrict__ aD,
                                const float* __restrict__ Mx,
                                float* __restrict__ Sm, int heads) {
  int t = blockIdx.x * blockDim.x + threadIdx.x;
  if (t >= ET * heads) return;
  int e = t / heads, hh = t - e * heads;
  int s, d; edge_src_dst(ei, E, e, s, d);
  float l = leaky(aS[s * heads + hh] + aD[d * heads + hh]);
  float ev = expf(l - Mx[d * heads + hh]);
  atomicAdd(&Sm[d * heads + hh], ev);
}

// --------------------------------------------- edge pass 3: weighted scatter
// One wave per edge. FOUT==128: each lane owns 4 consecutive channels (single
// float4 gather of h[src], one head per lane). FOUT==32: one channel per lane.
template <int HEADS, int CH>
__global__ void edge_scatter_kernel(const long long* __restrict__ ei, int E, int ET,
                                    const float* __restrict__ aS,
                                    const float* __restrict__ aD,
                                    const float* __restrict__ Mx,
                                    const float* __restrict__ Sm,
                                    const float* __restrict__ H,
                                    float* __restrict__ out) {
  constexpr int FOUT = HEADS * CH;
  int warp = (blockIdx.x * blockDim.x + threadIdx.x) >> 5;
  int lane = threadIdx.x & 31;
  if (warp >= ET) return;
  int s, d; edge_src_dst(ei, E, warp, s, d);

  if constexpr (FOUT == 128) {
    const int c  = lane * 4;           // 4 consecutive channels, same head
    const int hh = lane / 8;           // (4*lane)/32
    float l     = leaky(aS[s * HEADS + hh] + aD[d * HEADS + hh]);
    float alpha = expf(l - Mx[d * HEADS + hh]) / Sm[d * HEADS + hh];
    float4 hv = *(const float4*)(H + s * FOUT + c);
    float* op = out + d * FOUT + c;
    atomicAdd(op + 0, hv.x * alpha);
    atomicAdd(op + 1, hv.y * alpha);
    atomicAdd(op + 2, hv.z * alpha);
    atomicAdd(op + 3, hv.w * alpha);
  } else {
    const int hh = lane / CH;          // HEADS==1 -> hh==0
    float l     = leaky(aS[s * HEADS + hh] + aD[d * HEADS + hh]);
    float alpha = expf(l - Mx[d * HEADS + hh]) / Sm[d * HEADS + hh];
    atomicAdd(&out[d * FOUT + lane], H[s * FOUT + lane] * alpha);
  }
}

// --------------------------------------------------- bias + activation
// act: 0 = ELU, 1 = sigmoid
__global__ void finalize_kernel(const float* __restrict__ acc,
                                const float* __restrict__ bias,
                                float* __restrict__ dst,
                                int total, int fout, int act) {
  int i = blockIdx.x * blockDim.x + threadIdx.x;
  if (i >= total) return;
  int c = i % fout;
  float v = acc[i] + bias[c];
  if (act == 0) v = v > 0.0f ? v : (expf(v) - 1.0f);
  else          v = 1.0f / (1.0f + expf(-v));
  dst[i] = v;
}

// ------------------------------------------------------------------- host

extern "C" void kernel_launch(void* const* d_in, const int* in_sizes, int n_in,
                              void* d_out, int out_size, void* d_ws, size_t ws_size,
                              hipStream_t stream) {
  const float*     x  = (const float*)d_in[0];
  const long long* ei = (const long long*)d_in[1];
  const int N  = in_sizes[0] / 128;     // F_IN = 128
  const int E  = in_sizes[1] / 2;       // edge_index is [2,E]
  const int ET = E + N;                 // with self-loops

  const float *W[4], *As[4], *Ad[4], *B[4];
  for (int l = 0; l < 4; ++l) {
    W[l]  = (const float*)d_in[2 + 4 * l];
    As[l] = (const float*)d_in[3 + 4 * l];
    Ad[l] = (const float*)d_in[4 + 4 * l];
    B[l]  = (const float*)d_in[5 + 4 * l];
  }

  // workspace layout (floats): H | Acc | Xcur | aS | aD | Mx | Sm
  float* ws   = (float*)d_ws;
  float* Hbuf = ws;
  float* Acc  = Hbuf + (size_t)N * 128;
  float* Xcur = Acc  + (size_t)N * 128;
  float* aS   = Xcur + (size_t)N * 128;
  float* aD   = aS + (size_t)N * 4;
  float* Mx   = aD + (size_t)N * 4;
  float* Sm   = Mx + (size_t)N * 4;

  const int heads_arr[4] = {4, 4, 4, 1};
  const int ch_arr[4]    = {32, 32, 32, 32};

  const float* cur_in = x;
  for (int l = 0; l < 4; ++l) {
    const int heads = heads_arr[l];
    const int ch    = ch_arr[l];
    const int fout  = heads * ch;

    // 1) dense GEMM (WMMA): grid over (row tiles / 4 waves, col tiles)
    {
      dim3 grid((N + 63) / 64, fout / 16);
      if (fout == 128)
        gemm_wmma_kernel<128, 128><<<grid, 128, 0, stream>>>(cur_in, W[l], Hbuf, N);
      else
        gemm_wmma_kernel<128, 32><<<grid, 128, 0, stream>>>(cur_in, W[l], Hbuf, N);
    }

    // 2) per-node attention scores
    const int nh = N * heads;
    att_kernel<<<(nh + 255) / 256, 256, 0, stream>>>(Hbuf, As[l], Ad[l], aS, aD,
                                                     N, heads, ch);

    // 3) init segment buffers
    fill_kernel<<<(nh + 255) / 256, 256, 0, stream>>>(Mx, -INFINITY, nh);
    fill_kernel<<<(nh + 255) / 256, 256, 0, stream>>>(Sm, 0.0f, nh);
    const int nf = N * fout;
    fill_kernel<<<(nf + 255) / 256, 256, 0, stream>>>(Acc, 0.0f, nf);

    // 4) segment softmax over edges
    const int eth = ET * heads;
    edge_max_kernel<<<(eth + 255) / 256, 256, 0, stream>>>(ei, E, ET, aS, aD, Mx, heads);
    edge_sum_kernel<<<(eth + 255) / 256, 256, 0, stream>>>(ei, E, ET, aS, aD, Mx, Sm, heads);

    // 5) weighted message scatter: one wave per edge (8 waves per block)
    if (fout == 128)
      edge_scatter_kernel<4, 32><<<(ET + 7) / 8, 256, 0, stream>>>(ei, E, ET, aS, aD,
                                                                   Mx, Sm, Hbuf, Acc);
    else
      edge_scatter_kernel<1, 32><<<(ET + 7) / 8, 256, 0, stream>>>(ei, E, ET, aS, aD,
                                                                   Mx, Sm, Hbuf, Acc);

    // 6) bias + activation
    float* dstp = (l == 3) ? (float*)d_out : Xcur;
    finalize_kernel<<<(nf + 255) / 256, 256, 0, stream>>>(Acc, B[l], dstp, nf, fout,
                                                          (l == 3) ? 1 : 0);
    cur_in = Xcur;
  }
}
